// CustomMultiHeadAttention_43825846288415
// MI455X (gfx1250) — matrix operand order repaired and checked by reading, not compiled, not run on hardware
//
#include <hip/hip_runtime.h>

typedef _Float16 h16;
typedef __attribute__((ext_vector_type(16))) _Float16 v16h;
typedef __attribute__((ext_vector_type(8)))  _Float16 v8h;
typedef __attribute__((ext_vector_type(8)))  float    v8f;

#define HID 768
#define NH  12
#define HD  64
#define BB  2
#define SS  4096
#define BS  (BB*SS)            // 8192 rows
#define SCALE_INV 0.125f       // 1/sqrt(64)

// ---- WMMA fragment loaders (wave32, v_wmma_f32_16x16x32_f16) -------------
// A 16x32 (f16): lane L<16 holds row L, K = {0..7} U {16..23}; lane L>=16 row L-16, K = {8..15} U {24..31}
__device__ __forceinline__ v16h afrag_f32(const float* __restrict__ src, int ld, int row, int kb, int hl) {
  const float* p = src + (size_t)row * ld + kb + hl * 8;
  v16h f;
#pragma unroll
  for (int i = 0; i < 8; ++i) { f[i] = (h16)p[i]; f[8 + i] = (h16)p[16 + i]; }
  return f;
}
__device__ __forceinline__ v16h afrag_f16(const h16* __restrict__ src, int ld, int row, int kb, int hl) {
  const h16* p = src + (size_t)row * ld + kb + hl * 8;
  v16h f;
#pragma unroll
  for (int i = 0; i < 8; ++i) { f[i] = p[i]; f[8 + i] = p[16 + i]; }
  return f;
}
// B 32x16 (f16): lane holds column (lane%16); lanes 0-15 carry K = {0..7} U {16..23}, lanes 16-31 K = {8..15} U {24..31}
__device__ __forceinline__ v16h bfrag_f32(const float* __restrict__ src, int ld, int row, int kb, int hl) {
  const float* p = src + (size_t)row * ld + kb + hl * 8;
  v16h f;
#pragma unroll
  for (int i = 0; i < 8; ++i) { f[i] = (h16)p[i]; f[8 + i] = (h16)p[16 + i]; }
  return f;
}
__device__ __forceinline__ v16h bfrag_f16(const h16* __restrict__ src, int ld, int row, int kb, int hl) {
  const h16* p = src + (size_t)row * ld + kb + hl * 8;
  v16h f;
#pragma unroll
  for (int i = 0; i < 8; ++i) { f[i] = p[i]; f[8 + i] = p[16 + i]; }
  return f;
}
#define WMMA(a, b, c) __builtin_amdgcn_wmma_f32_16x16x32_f16(false, (a), false, (b), (short)0, (c), false, false)

// ---- Kernel 1: fused QKV projection, X[8192x768] x {Wq,Wk,Wv}^T + bias ----
// one 16x16 output tile per wave; Q,K stored f16 [b,h,s,d]; V stored f16 transposed [b,h,d,s]
__global__ __launch_bounds__(256) void qkv_kernel(const float* __restrict__ X,
    const float* __restrict__ Wq, const float* __restrict__ bq,
    const float* __restrict__ Wk, const float* __restrict__ bk,
    const float* __restrict__ Wv, const float* __restrict__ bv,
    h16* __restrict__ qf, h16* __restrict__ kf, h16* __restrict__ vt) {
  int wave = (blockIdx.x * 256 + threadIdx.x) >> 5;
  int lane = threadIdx.x & 31;
  int hl = lane >> 4, c = lane & 15;
  const int NT = 3 * HID / 16;           // 144 column tiles across Q|K|V
  int mtile = wave / NT;
  int ntile = wave - mtile * NT;
  int n0 = ntile * 16;
  int mat = n0 / HID;                    // 0=Q 1=K 2=V (wave-uniform)
  int wr = n0 - mat * HID;
  const float* W    = (mat == 0) ? Wq : (mat == 1) ? Wk : Wv;
  const float* bias = (mat == 0) ? bq : (mat == 1) ? bk : bv;
  int arow = mtile * 16 + c;
  int brow = wr + c;                     // B column c = row (wr+c) of W (since B = W^T)
  v8f acc = {};
  for (int kc = 0; kc < HID; kc += 32) {
    v16h a  = afrag_f32(X, HID, arow, kc, hl);
    v16h bf = bfrag_f32(W, HID, brow, kc, hl);
    acc = WMMA(a, bf, acc);
  }
  int ncol = wr + c;
  float bval = bias[ncol];
  int hh = ncol / HD, dd = ncol - hh * HD;
  int m0 = mtile * 16;                   // whole tile stays in one batch (4096 % 16 == 0)
  int bidx = m0 >> 12;
  int s0 = m0 & (SS - 1);
  // C layout: this lane owns rows m0 + hl*8 + r (r=0..7), column ncol.
  if (mat == 2) {
    // transposed V: [b,h,d,s] -> the 8 rows map to 8 CONSECUTIVE s -> one b128 store
    h16* p = vt + ((size_t)((bidx * NH + hh) * HD + dd)) * SS + s0 + hl * 8;
    v8h pk;
#pragma unroll
    for (int r = 0; r < 8; ++r) pk[r] = (h16)(acc[r] + bval);
    *reinterpret_cast<v8h*>(p) = pk;     // 16B-aligned by construction
  } else {
    h16* dst = (mat == 0) ? qf : kf;     // wave-uniform select, no per-store branching
    h16* p = dst + ((size_t)((bidx * NH + hh) * SS + s0 + hl * 8)) * HD + dd;
#pragma unroll
    for (int r = 0; r < 8; ++r) p[(size_t)r * HD] = (h16)(acc[r] + bval);
  }
}

// ---- Kernel 2: attention. One wave per 16 query rows of one (b,h). -------
// Pass 1: online softmax stats over all 4096 keys. Pass 2: recompute scores,
// write normalized fp32 attn weights (the 1.6GB output), LDS-transpose the
// f16 weights into A-fragments, accumulate context = attn @ V via WMMA.
__global__ __launch_bounds__(32) void attn_kernel(const h16* __restrict__ qf,
    const h16* __restrict__ kf, const h16* __restrict__ vt,
    const int* __restrict__ msk, float* __restrict__ attnout, h16* __restrict__ ctx) {
  __shared__ h16 lds[16 * 32];           // 16 queries x 32 keys f16 bounce tile
  int lane = threadIdx.x;
  int hl = lane >> 4, c = lane & 15;
  int qtile = blockIdx.x & (SS / 16 - 1);
  int bh = blockIdx.x >> 8;              // 0..23
  int b = bh / NH, h = bh % NH;
  const h16* Qb = qf + (size_t)bh * SS * HD;
  const h16* Kb = kf + (size_t)bh * SS * HD;
  const h16* Vb = vt + (size_t)bh * HD * SS;   // V transposed: row d, contiguous in s
  int q0 = qtile * 16;
  v16h qa0 = afrag_f16(Qb, HD, q0 + c, 0, hl);   // Q rows, d=0..31
  v16h qa1 = afrag_f16(Qb, HD, q0 + c, 32, hl);  // d=32..63

  float rmax[8], rsum[8];
#pragma unroll
  for (int r = 0; r < 8; ++r) { rmax[r] = -3.0e38f; rsum[r] = 0.f; }

  // ---- pass 1: stats ----
  for (int nt = 0; nt < SS / 16; ++nt) {
    if (nt + 8 < SS / 16)                // stage K tile ~8 iters ahead (global_prefetch_b8)
      __builtin_prefetch(Kb + ((size_t)(nt + 8) * 16) * HD + (size_t)lane * HD, 0, 1);
    v8f acc = {};
    v16h kb0 = bfrag_f16(Kb, HD, nt * 16 + c, 0, hl);   // B col c = key row
    v16h kb1 = bfrag_f16(Kb, HD, nt * 16 + c, 32, hl);
    acc = WMMA(qa0, kb0, acc);
    acc = WMMA(qa1, kb1, acc);
    bool dead = (msk[b * SS + nt * 16 + c] == 0);
#pragma unroll
    for (int r = 0; r < 8; ++r) {
      float s = dead ? -10000.f : acc[r] * SCALE_INV;
      float mn = fmaxf(rmax[r], s);
      rsum[r] = rsum[r] * __expf(rmax[r] - mn) + __expf(s - mn);
      rmax[r] = mn;
    }
  }
  // combine stats across the 16 lanes that share each row group
#pragma unroll
  for (int off = 1; off < 16; off <<= 1) {
#pragma unroll
    for (int r = 0; r < 8; ++r) {
      float om = __shfl_xor(rmax[r], off, 32);
      float os = __shfl_xor(rsum[r], off, 32);
      float mn = fmaxf(rmax[r], om);
      rsum[r] = rsum[r] * __expf(rmax[r] - mn) + os * __expf(om - mn);
      rmax[r] = mn;
    }
  }
  float rinv[8];
#pragma unroll
  for (int r = 0; r < 8; ++r) rinv[r] = (rsum[r] > 0.f) ? 1.f / rsum[r] : 0.f;

  // ---- pass 2: recompute, emit weights, accumulate context ----
  v8f ctxa[4] = {{}, {}, {}, {}};
  for (int nt2 = 0; nt2 < SS / 32; ++nt2) {
    if (nt2 * 2 + 8 < SS / 16)
      __builtin_prefetch(Kb + ((size_t)(nt2 * 2 + 8) * 16) * HD + (size_t)lane * HD, 0, 1);
#pragma unroll
    for (int sub = 0; sub < 2; ++sub) {
      int nt = nt2 * 2 + sub;
      v8f acc = {};
      v16h kb0 = bfrag_f16(Kb, HD, nt * 16 + c, 0, hl);
      v16h kb1 = bfrag_f16(Kb, HD, nt * 16 + c, 32, hl);
      acc = WMMA(qa0, kb0, acc);
      acc = WMMA(qa1, kb1, acc);
      bool dead = (msk[b * SS + nt * 16 + c] == 0);
#pragma unroll
      for (int r = 0; r < 8; ++r) {
        float s = dead ? -10000.f : acc[r] * SCALE_INV;
        float w = __expf(s - rmax[r]) * rinv[r];
        int row = hl * 8 + r;
        attnout[((size_t)(bh * SS + q0 + row)) * SS + nt * 16 + c] = w;
        lds[row * 32 + sub * 16 + c] = (h16)w;      // C-layout -> LDS
      }
    }
    __syncthreads();                                // single-wave WG: lowers to s_nop
    v16h af;                                        // LDS -> A-layout (transpose bounce)
#pragma unroll
    for (int i = 0; i < 8; ++i) {
      af[i]     = lds[c * 32 + hl * 8 + i];
      af[8 + i] = lds[c * 32 + 16 + hl * 8 + i];
    }
    __syncthreads();
#pragma unroll
    for (int dt = 0; dt < 4; ++dt) {
      v16h vb = bfrag_f16(Vb, SS, dt * 16 + c, nt2 * 32, hl);  // B col c = Vt row (d)
      ctxa[dt] = WMMA(af, vb, ctxa[dt]);
    }
  }
#pragma unroll
  for (int dt = 0; dt < 4; ++dt)
#pragma unroll
    for (int r = 0; r < 8; ++r) {
      int m = q0 + hl * 8 + r;
      ctx[((size_t)(b * SS + m)) * HID + h * HD + dt * 16 + c] = (h16)ctxa[dt][r];
    }
}

// ---- Kernel 3: out = ctx @ Wo^T + bo (fp32 output) -----------------------
__global__ __launch_bounds__(256) void proj_kernel(const h16* __restrict__ ctxp,
    const float* __restrict__ Wo, const float* __restrict__ bo, float* __restrict__ out) {
  int wave = (blockIdx.x * 256 + threadIdx.x) >> 5;
  int lane = threadIdx.x & 31;
  int hl = lane >> 4, c = lane & 15;
  const int NT = HID / 16;               // 48
  int mtile = wave / NT;
  int ntile = wave - mtile * NT;
  int arow = mtile * 16 + c;
  int brow = ntile * 16 + c;
  v8f acc = {};
  for (int kc = 0; kc < HID; kc += 32) {
    v16h a  = afrag_f16(ctxp, HID, arow, kc, hl);
    v16h bf = bfrag_f32(Wo, HID, brow, kc, hl);
    acc = WMMA(a, bf, acc);
  }
  float bval = bo[ntile * 16 + c];
#pragma unroll
  for (int r = 0; r < 8; ++r) {
    int m = mtile * 16 + hl * 8 + r;
    out[(size_t)m * HID + ntile * 16 + c] = acc[r] + bval;
  }
}

extern "C" void kernel_launch(void* const* d_in, const int* in_sizes, int n_in,
                              void* d_out, int out_size, void* d_ws, size_t ws_size,
                              hipStream_t stream) {
  const float* X  = (const float*)d_in[0];
  const int*   mk = (const int*)d_in[1];
  const float* Wq = (const float*)d_in[2];
  const float* bq = (const float*)d_in[3];
  const float* Wk = (const float*)d_in[4];
  const float* bk = (const float*)d_in[5];
  const float* Wv = (const float*)d_in[6];
  const float* bv = (const float*)d_in[7];
  const float* Wo = (const float*)d_in[8];
  const float* bo = (const float*)d_in[9];
  float* out = (float*)d_out;
  float* attnout = out + (size_t)BS * HID;       // second output region

  const size_t QKN = (size_t)BB * NH * SS * HD;  // 6,291,456 halfs each
  h16* ws  = (h16*)d_ws;                         // total: 4*QKN*2B = 50.3 MB
  h16* qf  = ws;
  h16* kf  = ws + QKN;
  h16* vt  = ws + 2 * QKN;
  h16* ctx = ws + 3 * QKN;

  {
    int waves = (BS / 16) * (3 * HID / 16);      // 73728 waves
    hipLaunchKernelGGL(qkv_kernel, dim3(waves / 8), dim3(256), 0, stream,
                       X, Wq, bq, Wk, bk, Wv, bv, qf, kf, vt);
  }
  hipLaunchKernelGGL(attn_kernel, dim3(BB * NH * (SS / 16)), dim3(32), 0, stream,
                     qf, kf, vt, mk, attnout, ctx);
  {
    int waves = (BS / 16) * (HID / 16);          // 24576 waves
    hipLaunchKernelGGL(proj_kernel, dim3(waves / 8), dim3(256), 0, stream,
                       ctx, Wo, bo, out);
  }
}
